// Seq2SeqModel_49529562857969
// MI455X (gfx1250) — compile-verified, matrix operand order checked by reading, not run on hardware
//
#include <hip/hip_runtime.h>
#include <hip/hip_bf16.h>

// Problem dims (from reference)
#define BB    256
#define TIN   128
#define TOUT  64
#define ND    128
#define HID   512
#define G3    1536   // 3*HID
#define PROJ  256
#define MT    16     // batch rows per workgroup (= WMMA M)
#define NWAVE 16
#define BLK   (NWAVE * 32)

typedef __attribute__((ext_vector_type(16))) __bf16       bf16x16;
typedef __attribute__((ext_vector_type(8)))  float        floatx8;
typedef __attribute__((ext_vector_type(4)))  unsigned int uintx4;

union FragU { bf16x16 v; uintx4 q[2]; };

__device__ inline unsigned short f2bf(float x) {
  unsigned int u = __float_as_uint(x);
  u = (u + 0x7FFFu + ((u >> 16) & 1u)) >> 16;   // round-to-nearest-even
  return (unsigned short)u;
}

__device__ inline float sigm(float x) { return 1.0f / (1.0f + __expf(-x)); }

// Load one 16x32 bf16 WMMA operand fragment (A or B) from a row-major
// [rows x rowlen] bf16 buffer (k contiguous). Per CDNA5 ISA 16-bit layout:
// lanes 0-15 hold K = kb..kb+7 (VGPR0-3) and kb+16..kb+23 (VGPR4-7),
// lanes 16-31 hold K = kb+8..kb+15 and kb+24..kb+31.  Two b128 loads.
__device__ inline bf16x16 load_frag(const unsigned short* base, int row,
                                    int rowlen, int kb, int half) {
  FragU f;
  const unsigned short* p = base + (size_t)row * rowlen + kb + half * 8;
  f.q[0] = *(const uintx4*)(p);
  f.q[1] = *(const uintx4*)(p + 16);
  return f.v;
}

__device__ inline floatx8 wmma_bf16(bf16x16 a, bf16x16 b, floatx8 c) {
  // (neg_a, A, neg_b, B, c_mod, C, reuse_a, reuse_b)
  return __builtin_amdgcn_wmma_f32_16x16x32_bf16(false, a, false, b,
                                                 (short)0, c, false, false);
}

// Fused GRU gate GEMM for one step:
//   G[:, 0:1024]   = x@Wi^T[rz] + h@Wh^T[rz] + (bi+bh)[rz]
//   G[:, 1024:1536]= x@Wi^T[n]  + bi[n]
//   Hn             = h@Wh^T[n]  + bh[n]
// 96 N-tiles split over 16 waves (6 each). Branches are wave-uniform.
__device__ inline void gru_gates_gemm(const unsigned short* Xb,
                                      const unsigned short* Hb,
                                      const unsigned short* __restrict__ Wi,
                                      const unsigned short* __restrict__ Wh,
                                      const float* Gbias, const float* Nbias,
                                      float* G, float* Hn) {
  const int lane = threadIdx.x & 31;
  const int wave = threadIdx.x >> 5;
  const int half = lane >> 4;
  const int col  = lane & 15;

  for (int ti = 0; ti < 6; ++ti) {
    const int tile = wave + ti * NWAVE;   // 0..95
    const int n0   = tile * 16;
    const int nrow = n0 + col;

    __builtin_prefetch(Wh + (size_t)nrow * HID, 0, 1);

    floatx8 acc = {};
#pragma unroll
    for (int kb = 0; kb < ND; kb += 32) {          // input GEMM, K=128
      bf16x16 a = load_frag(Xb, col, ND, kb, half);
      bf16x16 b = load_frag(Wi, nrow, ND, kb, half);
      acc = wmma_bf16(a, b, acc);
    }

    if (n0 < 2 * HID) {                            // r/z region: fuse gi+gh
#pragma unroll
      for (int kb = 0; kb < HID; kb += 32) {       // hidden GEMM, K=512
        bf16x16 a = load_frag(Hb, col, HID, kb, half);
        bf16x16 b = load_frag(Wh, nrow, HID, kb, half);
        acc = wmma_bf16(a, b, acc);
      }
      const float bias = Gbias[nrow];
#pragma unroll
      for (int v = 0; v < 8; ++v)
        G[(v + half * 8) * G3 + nrow] = acc[v] + bias;
    } else {                                       // n region: keep separate
      const float bias_i = Gbias[nrow];
#pragma unroll
      for (int v = 0; v < 8; ++v)
        G[(v + half * 8) * G3 + nrow] = acc[v] + bias_i;
      floatx8 acch = {};
#pragma unroll
      for (int kb = 0; kb < HID; kb += 32) {
        bf16x16 a = load_frag(Hb, col, HID, kb, half);
        bf16x16 b = load_frag(Wh, nrow, HID, kb, half);
        acch = wmma_bf16(a, b, acch);
      }
      const int j = nrow - 2 * HID;
      const float bias_h = Nbias[j];
#pragma unroll
      for (int v = 0; v < 8; ++v)
        Hn[(v + half * 8) * HID + j] = acch[v] + bias_h;
    }
  }
}

__global__ __launch_bounds__(BLK, 1)
void seq2seq_gru_wmma(const float* __restrict__ X,
                      const float* __restrict__ enc_bi, const float* __restrict__ enc_bh,
                      const float* __restrict__ dec_bi, const float* __restrict__ dec_bh,
                      const float* __restrict__ bp,     const float* __restrict__ bo,
                      const unsigned short* __restrict__ encWi,
                      const unsigned short* __restrict__ encWh,
                      const unsigned short* __restrict__ decWi,
                      const unsigned short* __restrict__ decWh,
                      const unsigned short* __restrict__ Wpw,
                      const unsigned short* __restrict__ Wow,
                      float* __restrict__ out) {
  extern __shared__ char smem[];
  unsigned short* Xbf = (unsigned short*)smem;       // 16x128 bf16 step input
  unsigned short* Hbf = Xbf + MT * ND;               // 16x512 bf16 hidden (WMMA A)
  unsigned short* Pbf = Hbf + MT * HID;              // 16x256 bf16 proj
  float* H32 = (float*)(Pbf + MT * PROJ);            // 16x512 f32 hidden (exact)
  float* G   = H32 + MT * HID;                       // 16x1536 f32 gates
  float* Hn  = G + MT * G3;                          // 16x512 f32 h_n part
  float* eGb = Hn + MT * HID;                        // 1536: enc bi(+bh on rz)
  float* eNb = eGb + G3;                             // 512 : enc bh[n]
  float* dGb = eNb + HID;                            // 1536
  float* dNb = dGb + G3;                             // 512
  float* bpL = dNb + HID;                            // 256
  float* boL = bpL + PROJ;                           // 128

  const int tid  = threadIdx.x;
  const int b0   = blockIdx.x * MT;
  const int lane = tid & 31;
  const int wave = tid >> 5;
  const int half = lane >> 4;
  const int col  = lane & 15;

  // One-time init: biases into LDS, h = 0
  for (int n = tid; n < G3; n += BLK) {
    eGb[n] = enc_bi[n] + (n < 2 * HID ? enc_bh[n] : 0.f);
    dGb[n] = dec_bi[n] + (n < 2 * HID ? dec_bh[n] : 0.f);
  }
  for (int j = tid; j < HID; j += BLK) {
    eNb[j] = enc_bh[2 * HID + j];
    dNb[j] = dec_bh[2 * HID + j];
  }
  for (int j = tid; j < PROJ; j += BLK) bpL[j] = bp[j];
  for (int j = tid; j < ND;   j += BLK) boL[j] = bo[j];
  for (int i = tid; i < MT * HID; i += BLK) { H32[i] = 0.f; Hbf[i] = 0; }
  __syncthreads();

  // ---------------- Encoder: 128 steps ----------------
  for (int t = 0; t < TIN; ++t) {
    for (int i = tid; i < MT * ND; i += BLK) {
      int m = i >> 7, c = i & (ND - 1);
      Xbf[i] = f2bf(X[((size_t)(b0 + m) * TIN + t) * ND + c]);
    }
    __syncthreads();
    gru_gates_gemm(Xbf, Hbf, encWi, encWh, eGb, eNb, G, Hn);
    __syncthreads();
    for (int i = tid; i < MT * HID; i += BLK) {
      int m = i >> 9, j = i & (HID - 1);
      float r  = sigm(G[m * G3 + j]);
      float z  = sigm(G[m * G3 + HID + j]);
      float nn = tanhf(G[m * G3 + 2 * HID + j] + r * Hn[i]);
      float h  = (1.f - z) * nn + z * H32[i];
      H32[i] = h;
      Hbf[i] = f2bf(h);
    }
    __syncthreads();
  }

  // Xbf now holds X[:, TIN-1, :] == decoder init input.
  // ---------------- Decoder: 64 steps ----------------
  for (int t = 0; t < TOUT; ++t) {
    gru_gates_gemm(Xbf, Hbf, decWi, decWh, dGb, dNb, G, Hn);
    __syncthreads();
    for (int i = tid; i < MT * HID; i += BLK) {
      int m = i >> 9, j = i & (HID - 1);
      float r  = sigm(G[m * G3 + j]);
      float z  = sigm(G[m * G3 + HID + j]);
      float nn = tanhf(G[m * G3 + 2 * HID + j] + r * Hn[i]);
      float h  = (1.f - z) * nn + z * H32[i];
      H32[i] = h;
      Hbf[i] = f2bf(h);
    }
    __syncthreads();

    // Projection: relu(h @ Wp^T + bp) -> Pbf (16x256), 16 tiles / 16 waves
    {
      const int n0 = wave * 16;
      const int nrow = n0 + col;
      floatx8 acc = {};
#pragma unroll
      for (int kb = 0; kb < HID; kb += 32) {
        bf16x16 a = load_frag(Hbf, col, HID, kb, half);
        bf16x16 b = load_frag(Wpw, nrow, HID, kb, half);
        acc = wmma_bf16(a, b, acc);
      }
      const float bias = bpL[nrow];
#pragma unroll
      for (int v = 0; v < 8; ++v) {
        float p = acc[v] + bias;
        p = p > 0.f ? p : 0.f;
        Pbf[(v + half * 8) * PROJ + nrow] = f2bf(p);
      }
    }
    __syncthreads();

    // Output head: p @ Wo^T + bo -> out (16x128), 8 tiles / waves 0..7
    if (wave < 8) {
      const int n0 = wave * 16;
      const int nrow = n0 + col;
      floatx8 acc = {};
#pragma unroll
      for (int kb = 0; kb < PROJ; kb += 32) {
        bf16x16 a = load_frag(Pbf, col, PROJ, kb, half);
        bf16x16 b = load_frag(Wow, nrow, PROJ, kb, half);
        acc = wmma_bf16(a, b, acc);
      }
      const float bias = boL[nrow];
#pragma unroll
      for (int v = 0; v < 8; ++v) {
        int m = v + half * 8;
        float o = acc[v] + bias;
        out[((size_t)(b0 + m) * TOUT + t) * ND + nrow] = o;
        Xbf[m * ND + nrow] = f2bf(o);   // feed back as next decoder input
      }
    }
    __syncthreads();
  }
}

__global__ void cvt_f32_bf16(const float* __restrict__ s,
                             unsigned short* __restrict__ d, int n) {
  int i = blockIdx.x * blockDim.x + threadIdx.x;
  if (i < n) d[i] = f2bf(s[i]);
}

extern "C" void kernel_launch(void* const* d_in, const int* in_sizes, int n_in,
                              void* d_out, int out_size, void* d_ws, size_t ws_size,
                              hipStream_t stream) {
  (void)in_sizes; (void)n_in; (void)out_size; (void)ws_size;
  const float* X      = (const float*)d_in[0];
  const float* enc_Wi = (const float*)d_in[1];
  const float* enc_Wh = (const float*)d_in[2];
  const float* enc_bi = (const float*)d_in[3];
  const float* enc_bh = (const float*)d_in[4];
  const float* dec_Wi = (const float*)d_in[5];
  const float* dec_Wh = (const float*)d_in[6];
  const float* dec_bi = (const float*)d_in[7];
  const float* dec_bh = (const float*)d_in[8];
  const float* Wp     = (const float*)d_in[9];
  const float* bp     = (const float*)d_in[10];
  const float* Wo     = (const float*)d_in[11];
  const float* bo     = (const float*)d_in[12];

  // Workspace layout: bf16 copies of all weights (~4.1 MB), L2-resident.
  unsigned short* ws       = (unsigned short*)d_ws;
  unsigned short* encWi_bf = ws;
  unsigned short* encWh_bf = encWi_bf + G3 * ND;
  unsigned short* decWi_bf = encWh_bf + G3 * HID;
  unsigned short* decWh_bf = decWi_bf + G3 * ND;
  unsigned short* Wp_bf    = decWh_bf + G3 * HID;
  unsigned short* Wo_bf    = Wp_bf    + PROJ * HID;

  auto cvt = [&](const float* s, unsigned short* d, int n) {
    cvt_f32_bf16<<<(n + 255) / 256, 256, 0, stream>>>(s, d, n);
  };
  cvt(enc_Wi, encWi_bf, G3 * ND);
  cvt(enc_Wh, encWh_bf, G3 * HID);
  cvt(dec_Wi, decWi_bf, G3 * ND);
  cvt(dec_Wh, decWh_bf, G3 * HID);
  cvt(Wp,     Wp_bf,    PROJ * HID);
  cvt(Wo,     Wo_bf,    PROJ * ND);

  const size_t smem =
      (size_t)(MT * ND + MT * HID + MT * PROJ) * 2 +                 // bf16 bufs
      (size_t)(MT * HID + MT * G3 + MT * HID +                       // f32 bufs
               G3 + HID + G3 + HID + PROJ + ND) * 4;                 // biases

  seq2seq_gru_wmma<<<BB / MT, BLK, smem, stream>>>(
      X, enc_bi, enc_bh, dec_bi, dec_bh, bp, bo,
      encWi_bf, encWh_bf, decWi_bf, decWh_bf, Wp_bf, Wo_bf,
      (float*)d_out);
}
